// MultiHeadSelfAttention_29961691857644
// MI455X (gfx1250) — matrix operand order repaired and checked by reading, not compiled, not run on hardware
//
#include <hip/hip_runtime.h>
#include <hip/hip_bf16.h>

// MultiHeadSelfAttention for MI455X (gfx1250), bf16 WMMA path.
// B=8, S=1024, D=768, H=12, DH=64. scale = 1/sqrt(768) folded into Q.

#define B_   8
#define S_   1024
#define D_   768
#define H_   12
#define DH_  64
#define M_   (B_ * S_)                     // 8192 tokens
#define SCALE_ 0.03608439182435161f        // 1/sqrt(768)

typedef __attribute__((ext_vector_type(16))) __bf16 v16bf;
typedef __attribute__((ext_vector_type(8)))  float  v8f;

union BFrag {
  v16bf v;
  uint4 q[2];
};

__device__ __forceinline__ unsigned short f2bf(float f) {
  // round-to-nearest-even fp32 -> bf16
  unsigned u = __float_as_uint(f);
  u += 0x7fffu + ((u >> 16) & 1u);
  return (unsigned short)(u >> 16);
}

__device__ __forceinline__ v8f wmma_bf16(const BFrag& a, const BFrag& b, v8f c) {
  // (neg_a, A, neg_b, B, c_mod, C, reuse_a, reuse_b)
  return __builtin_amdgcn_wmma_f32_16x16x32_bf16(false, a.v, false, b.v,
                                                 (short)0, c, false, false);
}

// ---------------------------------------------------------------------------
// Kernel 1: x (fp32) -> bf16
// ---------------------------------------------------------------------------
__global__ void cvt_x_kernel(const float* __restrict__ x,
                             unsigned short* __restrict__ xb, int n) {
  int i = blockIdx.x * 256 + threadIdx.x;
  if (i < n) xb[i] = f2bf(x[i]);
}

// ---------------------------------------------------------------------------
// Kernel 2: Wq/Wk/Wv (fp32, [k][n]) -> wt bf16 [w][n][k]  (transposed so that
// WMMA B-fragments read two runs of 8 contiguous k-values for a fixed output column n)
// ---------------------------------------------------------------------------
__global__ void cvt_w_kernel(const float* __restrict__ Wq,
                             const float* __restrict__ Wk,
                             const float* __restrict__ Wv,
                             unsigned short* __restrict__ wt) {
  int i = blockIdx.x * 256 + threadIdx.x;
  if (i >= 3 * D_ * D_) return;
  int w = i / (D_ * D_);
  int r = i % (D_ * D_);
  int n = r / D_;
  int k = r % D_;
  const float* W = (w == 0) ? Wq : (w == 1) ? Wk : Wv;
  wt[i] = f2bf(W[k * D_ + n]);
}

// ---------------------------------------------------------------------------
// Kernel 3: QKV projection GEMM.  grid = (M/128, D/128, 3), block = 256
// Block tile 128x128, K-step 32 through LDS.  8 waves as 4(M) x 2(N); each
// wave owns a 32x64 tile = 2x4 grid of 16x16 WMMA tiles (8 WMMA / K-step
// from 6 LDS fragment loads).  Epilogue: +bias (and *SCALE for Q); store
// Q,K as bf16 [b][h][s][dh], V transposed as bf16 [b][h][dh][s].
// ---------------------------------------------------------------------------
#define ASTR 40   // LDS row stride (bf16 elems), multiple of 8, padded
#define BSTR 40

__global__ __launch_bounds__(256) void qkv_gemm_kernel(
    const unsigned short* __restrict__ xb,   // [8192][768] bf16
    const unsigned short* __restrict__ wt,   // [3][768(n)][768(k)] bf16
    const float* __restrict__ bq,
    const float* __restrict__ bk,
    const float* __restrict__ bv,
    unsigned short* __restrict__ Qb,         // [b][h][s][dh] (scaled)
    unsigned short* __restrict__ Kb,         // [b][h][s][dh]
    unsigned short* __restrict__ Vt)         // [b][h][dh][s]
{
  __shared__ __align__(16) unsigned short As[128 * ASTR];
  __shared__ __align__(16) unsigned short Bs[128 * BSTR];

  const int t    = threadIdx.x;
  const int lane = t & 31;
  const int wave = t >> 5;
  const int col  = lane & 15;   // tile row (A) / column (B)
  const int half = lane >> 4;   // lane half selects K sub-range
  const int m0   = blockIdx.x * 128;
  const int n0   = blockIdx.y * 128;
  const int w    = blockIdx.z;  // 0=Q 1=K 2=V
  const int mOff = (wave & 3) * 32;
  const int nOff = (wave >> 2) * 64;

  const unsigned short* wmat = wt + (size_t)w * D_ * D_;

  // cooperative-load indices: 512 x 16B chunks per tile, 2 chunks/thread
  const int ldRow = (t * 2) >> 2;        // 0..127  (both chunks same row)
  const int ldCk0 = (t * 2) & 3;         // chunk 0 within row
  const int ldCk1 = ldCk0 + 1;

  v8f acc[2][4];
#pragma unroll
  for (int mi = 0; mi < 2; ++mi)
#pragma unroll
    for (int ni = 0; ni < 4; ++ni)
#pragma unroll
      for (int r = 0; r < 8; ++r) acc[mi][ni][r] = 0.0f;

  for (int k0 = 0; k0 < D_; k0 += 32) {
    // stage A: 128 rows x 32 bf16
    {
      const unsigned short* g = xb + (size_t)(m0 + ldRow) * D_ + k0;
      *(uint4*)(As + ldRow * ASTR + ldCk0 * 8) = *(const uint4*)(g + ldCk0 * 8);
      *(uint4*)(As + ldRow * ASTR + ldCk1 * 8) = *(const uint4*)(g + ldCk1 * 8);
      if (k0 + 32 < D_) __builtin_prefetch(g + 32, 0, 1);
    }
    // stage B: 128 rows(n) x 32 bf16(k)
    {
      const unsigned short* g = wmat + (size_t)(n0 + ldRow) * D_ + k0;
      *(uint4*)(Bs + ldRow * BSTR + ldCk0 * 8) = *(const uint4*)(g + ldCk0 * 8);
      *(uint4*)(Bs + ldRow * BSTR + ldCk1 * 8) = *(const uint4*)(g + ldCk1 * 8);
      if (k0 + 32 < D_) __builtin_prefetch(g + 32, 0, 1);
    }
    __syncthreads();

    // A fragment (16x32): row = col, chunks at K = half*8 and 16+half*8
    BFrag a[2], b[4];
#pragma unroll
    for (int mi = 0; mi < 2; ++mi) {
      const unsigned short* p = As + (mOff + mi * 16 + col) * ASTR;
      a[mi].q[0] = *(const uint4*)(p + half * 8);
      a[mi].q[1] = *(const uint4*)(p + 16 + half * 8);
    }
    // B fragment (32x16): column = col, chunks at K = half*8 and 16+half*8
#pragma unroll
    for (int ni = 0; ni < 4; ++ni) {
      const unsigned short* p = Bs + (nOff + ni * 16 + col) * BSTR;
      b[ni].q[0] = *(const uint4*)(p + half * 8);
      b[ni].q[1] = *(const uint4*)(p + 16 + half * 8);
    }
#pragma unroll
    for (int mi = 0; mi < 2; ++mi)
#pragma unroll
      for (int ni = 0; ni < 4; ++ni)
        acc[mi][ni] = wmma_bf16(a[mi], b[ni], acc[mi][ni]);
    __syncthreads();
  }

  // Epilogue.  C layout: element r of lane -> row = half*8 + r, col = col.
  const float* bias = (w == 0) ? bq : (w == 1) ? bk : bv;
  const float  sc   = (w == 0) ? SCALE_ : 1.0f;
#pragma unroll
  for (int mi = 0; mi < 2; ++mi) {
#pragma unroll
    for (int ni = 0; ni < 4; ++ni) {
      int ng = n0 + nOff + ni * 16 + col;     // global output column 0..767
      int hh = ng >> 6;
      int dh = ng & 63;
      float bb = bias[ng];
#pragma unroll
      for (int r = 0; r < 8; ++r) {
        int mg = m0 + mOff + mi * 16 + half * 8 + r;   // global token 0..8191
        int bi = mg >> 10;
        int ss = mg & 1023;
        unsigned short o = f2bf((acc[mi][ni][r] + bb) * sc);
        size_t head = (size_t)bi * H_ + hh;
        if (w == 0)      Qb[(head * S_ + ss) * DH_ + dh] = o;
        else if (w == 1) Kb[(head * S_ + ss) * DH_ + dh] = o;
        else             Vt[(head * DH_ + dh) * S_ + ss] = o;
      }
    }
  }
}

// ---------------------------------------------------------------------------
// Kernel 4: flash attention.  grid = B*H*(S/256) = 384 blocks, block = 256.
// Each wave owns 32 query rows (two 16-row tiles); 8 waves share 32-key K/V
// tiles via LDS.  K/V fragments are loaded once per key-step and reused by
// both query tiles: 16 WMMA per key-step.
// ---------------------------------------------------------------------------
__global__ __launch_bounds__(256) void flash_attn_kernel(
    const unsigned short* __restrict__ Qb,   // [b][h][s][dh], pre-scaled
    const unsigned short* __restrict__ Kb,   // [b][h][s][dh]
    const unsigned short* __restrict__ Vt,   // [b][h][dh][s]
    float* __restrict__ out)                 // [b][s][d] fp32
{
  __shared__ __align__(16) unsigned short Ks[32 * 64];     // [key][dh]
  __shared__ __align__(16) unsigned short Vs[64 * 32];     // [dh][key]
  __shared__ __align__(16) unsigned short Ps[8][16 * 32];  // per-wave P tile

  const int t    = threadIdx.x;
  const int lane = t & 31;
  const int wave = t >> 5;
  const int col  = lane & 15;
  const int half = lane >> 4;

  const int qg = blockIdx.x & 3;
  const int h  = (blockIdx.x >> 2) % H_;
  const int b  = blockIdx.x / (4 * H_);
  const int s0 = qg * 256 + wave * 32;      // this wave's 32 query rows

  const size_t head = (size_t)b * H_ + h;
  const unsigned short* qp = Qb + head * S_ * DH_;
  const unsigned short* kp = Kb + head * S_ * DH_;
  const unsigned short* vp = Vt + head * DH_ * S_;

  // Q as A-fragments: [query tile][dh 0..31 | 32..63], loaded once.
  BFrag aq[2][2];
#pragma unroll
  for (int qt = 0; qt < 2; ++qt) {
    const unsigned short* qrow = qp + (size_t)(s0 + qt * 16 + col) * DH_;
    aq[qt][0].q[0] = *(const uint4*)(qrow + half * 8);
    aq[qt][0].q[1] = *(const uint4*)(qrow + 16 + half * 8);
    aq[qt][1].q[0] = *(const uint4*)(qrow + 32 + half * 8);
    aq[qt][1].q[1] = *(const uint4*)(qrow + 48 + half * 8);
  }

  v8f o[2][4];
  float mrow[2][8], lrow[2][8];
#pragma unroll
  for (int qt = 0; qt < 2; ++qt) {
#pragma unroll
    for (int j = 0; j < 4; ++j)
#pragma unroll
      for (int r = 0; r < 8; ++r) o[qt][j][r] = 0.0f;
#pragma unroll
    for (int r = 0; r < 8; ++r) { mrow[qt][r] = -INFINITY; lrow[qt][r] = 0.0f; }
  }

  // cooperative-load indices for K (8 chunks/row) and V (4 chunks/row)
  const int kRow = t >> 3, kCk = t & 7;
  const int vRow = t >> 2, vCk = t & 3;

  for (int kk = 0; kk < S_; kk += 32) {
    // stage K tile [32 keys][64 dh] and V tile [64 dh][32 keys]
    {
      const unsigned short* g = kp + (size_t)(kk + kRow) * DH_ + kCk * 8;
      *(uint4*)(Ks + kRow * 64 + kCk * 8) = *(const uint4*)g;
      if (kk + 32 < S_) __builtin_prefetch(g + 32 * DH_, 0, 1);
    }
    {
      const unsigned short* g = vp + (size_t)vRow * S_ + kk + vCk * 8;
      *(uint4*)(Vs + vRow * 32 + vCk * 8) = *(const uint4*)g;
      if (kk + 32 < S_) __builtin_prefetch(g + 32, 0, 1);
    }
    __syncthreads();

    // K fragments (per 16-key tile, per dh-half) -- shared by both q-tiles
    BFrag bk[2][2];
#pragma unroll
    for (int kt = 0; kt < 2; ++kt) {
      const unsigned short* kr = Ks + (kt * 16 + col) * 64;
      bk[kt][0].q[0] = *(const uint4*)(kr + half * 8);
      bk[kt][0].q[1] = *(const uint4*)(kr + 16 + half * 8);
      bk[kt][1].q[0] = *(const uint4*)(kr + 32 + half * 8);
      bk[kt][1].q[1] = *(const uint4*)(kr + 48 + half * 8);
    }
    // V fragments (per 16-wide dh tile) -- shared by both q-tiles
    BFrag bv[4];
#pragma unroll
    for (int j = 0; j < 4; ++j) {
      const unsigned short* vr = Vs + (j * 16 + col) * 32;
      bv[j].q[0] = *(const uint4*)(vr + half * 8);
      bv[j].q[1] = *(const uint4*)(vr + 16 + half * 8);
    }

#pragma unroll
    for (int qt = 0; qt < 2; ++qt) {
      // Scores: two 16x16 tiles, contraction dh=64 in 2 WMMAs each.
      v8f scf[2];
#pragma unroll
      for (int kt = 0; kt < 2; ++kt) {
        v8f z = {0.f, 0.f, 0.f, 0.f, 0.f, 0.f, 0.f, 0.f};
        z = wmma_bf16(aq[qt][0], bk[kt][0], z);
        z = wmma_bf16(aq[qt][1], bk[kt][1], z);
        scf[kt] = z;
      }

      // Online softmax.  Lane's element r lives in query row (half*8 + r);
      // xor masks 1/2/4/8 reduce across the 16 lanes of that row.
#pragma unroll
      for (int r = 0; r < 8; ++r) {
        float v0 = scf[0][r], v1 = scf[1][r];
        float tm = fmaxf(v0, v1);
        tm = fmaxf(tm, __shfl_xor(tm, 1));
        tm = fmaxf(tm, __shfl_xor(tm, 2));
        tm = fmaxf(tm, __shfl_xor(tm, 4));
        tm = fmaxf(tm, __shfl_xor(tm, 8));
        float mn   = fmaxf(mrow[qt][r], tm);
        float corr = __expf(mrow[qt][r] - mn);
        float p0   = __expf(v0 - mn);
        float p1   = __expf(v1 - mn);
        float rs = p0 + p1;
        rs += __shfl_xor(rs, 1);
        rs += __shfl_xor(rs, 2);
        rs += __shfl_xor(rs, 4);
        rs += __shfl_xor(rs, 8);
        lrow[qt][r] = lrow[qt][r] * corr + rs;
        mrow[qt][r] = mn;
#pragma unroll
        for (int j = 0; j < 4; ++j) o[qt][j][r] *= corr;
        // stage P (bf16) per-wave for C->A fragment reshuffle
        Ps[wave][(half * 8 + r) * 32 + col]      = f2bf(p0);
        Ps[wave][(half * 8 + r) * 32 + 16 + col] = f2bf(p1);
      }

      // P as A-fragment (16x32 over key dim); same-wave LDS is in-order.
      BFrag ap;
      {
        const unsigned short* pr = &Ps[wave][col * 32];
        ap.q[0] = *(const uint4*)(pr + half * 8);
        ap.q[1] = *(const uint4*)(pr + 16 + half * 8);
      }
      // O += P x V : 4 column tiles of dh
#pragma unroll
      for (int j = 0; j < 4; ++j)
        o[qt][j] = wmma_bf16(ap, bv[j], o[qt][j]);
    }
    __syncthreads();
  }

  // Normalize and write fp32 output [b][s][h*64+dh]
#pragma unroll
  for (int qt = 0; qt < 2; ++qt) {
#pragma unroll
    for (int r = 0; r < 8; ++r) {
      float inv = 1.0f / lrow[qt][r];
      int s = s0 + qt * 16 + half * 8 + r;
#pragma unroll
      for (int j = 0; j < 4; ++j) {
        int c = h * DH_ + j * 16 + col;
        out[((size_t)b * S_ + s) * D_ + c] = o[qt][j][r] * inv;
      }
    }
  }
}

// ---------------------------------------------------------------------------
// Host launcher.  Workspace layout (bf16, 256B-aligned):
//   xb[8192*768] + wt[3*768*768] + Qb + Kb + Vt[8*12*1024*64]  ~= 54 MB
// ---------------------------------------------------------------------------
extern "C" void kernel_launch(void* const* d_in, const int* in_sizes, int n_in,
                              void* d_out, int out_size, void* d_ws, size_t ws_size,
                              hipStream_t stream) {
  const float* x  = (const float*)d_in[0];
  const float* Wq = (const float*)d_in[1];
  const float* bq = (const float*)d_in[2];
  const float* Wk = (const float*)d_in[3];
  const float* bk = (const float*)d_in[4];
  const float* Wv = (const float*)d_in[5];
  const float* bv = (const float*)d_in[6];
  float* out = (float*)d_out;

  char* ws = (char*)d_ws;
  size_t off = 0;
  auto take = [&](size_t elems) {
    unsigned short* p = (unsigned short*)(ws + off);
    off += ((elems * sizeof(unsigned short) + 255) & ~(size_t)255);
    return p;
  };
  unsigned short* xb = take((size_t)M_ * D_);
  unsigned short* wt = take((size_t)3 * D_ * D_);
  unsigned short* Qb = take((size_t)B_ * H_ * S_ * DH_);
  unsigned short* Kb = take((size_t)B_ * H_ * S_ * DH_);
  unsigned short* Vt = take((size_t)B_ * H_ * S_ * DH_);

  cvt_x_kernel<<<(M_ * D_ + 255) / 256, 256, 0, stream>>>(x, xb, M_ * D_);
  cvt_w_kernel<<<(3 * D_ * D_ + 255) / 256, 256, 0, stream>>>(Wq, Wk, Wv, wt);
  qkv_gemm_kernel<<<dim3(M_ / 128, D_ / 128, 3), 256, 0, stream>>>(
      xb, wt, bq, bk, bv, Qb, Kb, Vt);
  flash_attn_kernel<<<B_ * H_ * (S_ / 256), 256, 0, stream>>>(Qb, Kb, Vt, out);
}
